// Attention_60481729462939
// MI455X (gfx1250) — matrix operand order repaired and checked by reading, not compiled, not run on hardware
//
#include <hip/hip_runtime.h>

typedef __attribute__((ext_vector_type(16))) __bf16 bf16x16;
typedef __attribute__((ext_vector_type(8)))  __bf16 bf16x8;
typedef __attribute__((ext_vector_type(8)))  float  f32x8;

#define NB   16      // batch
#define NT   1024    // seq len
#define NC   768     // channels
#define NH   12      // heads
#define ND   64      // head dim
#define NEGV (-1e9f)

// ---- WMMA fragment layouts as used in this file (wave32) ----
// A (16x32 bf16): lane -> M = lane&15 ; elem j -> K = (lane>>4)*16 + j   (16 contiguous)
// B (32x16 bf16): lane -> N = lane&15 ; elem j -> K = (lane>>4)*16 + j   (16 contiguous)
// C/D (16x16 f32): lane -> N = lane&15 ; elem r -> M = (lane>>4)*8 + r

__device__ __forceinline__ bf16x16 load_a_frag(const __bf16* rowk0, int lane) {
  const int kb = (lane >> 4) * 16;
  const bf16x8 lo = *reinterpret_cast<const bf16x8*>(rowk0 + kb);
  const bf16x8 hi = *reinterpret_cast<const bf16x8*>(rowk0 + kb + 8);
  bf16x16 a;
#pragma unroll
  for (int i = 0; i < 8; ++i) { a[i] = lo[i]; a[i + 8] = hi[i]; }
  return a;
}

// ---------------- precision / layout prep ----------------
__global__ __launch_bounds__(256) void cvt_bf16_kernel(const float* __restrict__ src,
                                                       __bf16* __restrict__ dst, int n) {
  int i = blockIdx.x * 256 + threadIdx.x;
  if (i < n) dst[i] = (__bf16)src[i];
}

// Wt[n*K + k] = W[k*N + n]  (so WMMA B-fragments are contiguous along K)
__global__ __launch_bounds__(256) void transpose_cvt_kernel(const float* __restrict__ W,
                                                            __bf16* __restrict__ Wt,
                                                            int K, int N) {
  int i = blockIdx.x * 256 + threadIdx.x;
  if (i < K * N) {
    int n = i / K, k = i - n * K;
    Wt[i] = (__bf16)W[(size_t)k * N + n];
  }
}

// ---------------- QKV GEMM: [16384,768] @ Wt[2304,768]^T + bias ----------------
// one wave computes a 16(M) x 64(N) tile; epilogue scatters into Q/K ([B,H,T,D]) and Vt ([B,H,D,T])
__global__ __launch_bounds__(32) void qkv_gemm_kernel(const __bf16* __restrict__ xbf,
                                                      const __bf16* __restrict__ WtA,
                                                      const float* __restrict__ b_attn,
                                                      __bf16* __restrict__ Q,
                                                      __bf16* __restrict__ Kc,
                                                      __bf16* __restrict__ Vt) {
  const int lane = threadIdx.x & 31;
  const int m0 = blockIdx.x * 16;
  const int n0 = blockIdx.y * 64;
  const int m    = lane & 15;
  const int nn   = lane & 15;
  const int kb_b = (lane >> 4) * 16;
  f32x8 acc[4] = {};
  const __bf16* Arow = xbf + (size_t)(m0 + m) * NC;
  for (int k0 = 0; k0 < NC; k0 += 32) {
    bf16x16 a = load_a_frag(Arow + k0, lane);
#pragma unroll
    for (int t = 0; t < 4; ++t) {
      const __bf16* Bp = WtA + (size_t)(n0 + t * 16 + nn) * NC + k0 + kb_b;
      bf16x16 bfr = *reinterpret_cast<const bf16x16*>(Bp);
      acc[t] = __builtin_amdgcn_wmma_f32_16x16x32_bf16(false, a, false, bfr,
                                                       (short)0, acc[t], false, false);
    }
  }
  const int rbase = (lane >> 4) * 8;
#pragma unroll
  for (int t = 0; t < 4; ++t) {
    const int c = n0 + t * 16 + nn;
    const float bias = b_attn[c];
#pragma unroll
    for (int r = 0; r < 8; ++r) {
      const int row = m0 + rbase + r;
      const int bb = row >> 10, tok = row & (NT - 1);
      const float v = acc[t][r] + bias;
      if (c < NC) {                       // Q -> [B,H,T,D]
        const int h = c >> 6, d = c & 63;
        Q[(((size_t)(bb * NH + h)) * NT + tok) * ND + d] = (__bf16)v;
      } else if (c < 2 * NC) {            // K -> [B,H,T,D]
        const int c2 = c - NC, h = c2 >> 6, d = c2 & 63;
        Kc[(((size_t)(bb * NH + h)) * NT + tok) * ND + d] = (__bf16)v;
      } else {                            // V -> [B,H,D,T] (transposed for PV B-frags)
        const int c2 = c - 2 * NC, h = c2 >> 6, d = c2 & 63;
        Vt[(((size_t)(bb * NH + h)) * ND + d) * NT + tok] = (__bf16)v;
      }
    }
  }
}

// LDS score tile: 16 rows x 1024 cols f32 = 64KB; XOR swizzle on col bits 2..5 so the
// PV A-fragment read (16 lanes = 16 rows, same col) hits 16 distinct banks.
__device__ __forceinline__ int sidx(int row, int col) {
  return row * NT + (col ^ ((row & 15) << 2));
}

// ---------------- attention: one block = (b, h, 16-query tile), 4 waves ----------------
__global__ __launch_bounds__(128) void attn_kernel(const __bf16* __restrict__ Q,
                                                   const __bf16* __restrict__ Kc,
                                                   const __bf16* __restrict__ Vt,
                                                   const int* __restrict__ amask,
                                                   __bf16* __restrict__ Y) {
  __shared__ float S[16 * NT];   // 64KB
  const int qt = blockIdx.x, h = blockIdx.y, b = blockIdx.z;
  const int bh = b * NH + h;
  const int q0 = qt * 16;
  const int w = threadIdx.x >> 5, lane = threadIdx.x & 31;
  const int m = lane & 15, nn = lane & 15;
  const int rbase = (lane >> 4) * 8;
  const int kb_a = (lane >> 4) * 16;
  const int kb_b = (lane >> 4) * 16;
  const float scale = 0.125f;   // 1/sqrt(64)

  // ---- phase 1: S = scale * Q K^T with causal + key mask ----
  const __bf16* Qrow = Q + ((size_t)bh * NT + q0 + m) * ND;
  for (int kt = w; kt < NT / 16; kt += 4) {
    const int kc = kt * 16;
    f32x8 acc = {};
    if (kc <= q0 + 15) {   // tile not fully beyond causal bound
#pragma unroll
      for (int k0 = 0; k0 < ND; k0 += 32) {
        bf16x16 a = load_a_frag(Qrow + k0, lane);
        const __bf16* Kp = Kc + ((size_t)bh * NT + kc + nn) * ND + k0 + kb_b;
        bf16x16 bfr = *reinterpret_cast<const bf16x16*>(Kp);
        acc = __builtin_amdgcn_wmma_f32_16x16x32_bf16(false, a, false, bfr,
                                                      (short)0, acc, false, false);
      }
    }
    const int key = kc + nn;
    const bool kvalid = (amask[b * NT + key] != 0);
#pragma unroll
    for (int r = 0; r < 8; ++r) {
      const int row = rbase + r;
      const bool ok = kvalid && (key <= q0 + row);
      S[sidx(row, key)] = ok ? acc[r] * scale : NEGV;
    }
  }
  __syncthreads();

  // ---- phase 2: row softmax (8 threads per row, shuffle reductions) ----
  {
    const int row = threadIdx.x >> 3, sub = threadIdx.x & 7;
    float mx = -3.0e38f;
    for (int c = sub; c < NT; c += 8) mx = fmaxf(mx, S[sidx(row, c)]);
#pragma unroll
    for (int off = 4; off > 0; off >>= 1) mx = fmaxf(mx, __shfl_xor(mx, off, 8));
    float sum = 0.f;
    for (int c = sub; c < NT; c += 8) {
      const float e = __expf(S[sidx(row, c)] - mx);
      S[sidx(row, c)] = e;
      sum += e;
    }
#pragma unroll
    for (int off = 4; off > 0; off >>= 1) sum += __shfl_xor(sum, off, 8);
    const float qm = (amask[b * NT + q0 + row] != 0) ? 1.f : 0.f;
    const float inv = qm / sum;   // query mask folded into normalization
    for (int c = sub; c < NT; c += 8) S[sidx(row, c)] *= inv;
  }
  __syncthreads();

  // ---- phase 3: Y_tile = P @ V ; wave w owns D-block [w*16, w*16+16) ----
  const int d0 = w * 16;
  f32x8 acc = {};
  const int kend = min(NT, (q0 + 16 + 31) & ~31);   // probs beyond causal bound are ~0
  for (int kc = 0; kc < kend; kc += 32) {
    bf16x16 a;
#pragma unroll
    for (int j = 0; j < 8; ++j) {
      a[j]     = (__bf16)S[sidx(m, kc + kb_a + j)];
      a[j + 8] = (__bf16)S[sidx(m, kc + kb_a + 8 + j)];
    }
    const __bf16* Vp = Vt + ((size_t)bh * ND + d0 + nn) * NT + kc + kb_b;
    bf16x16 bfr = *reinterpret_cast<const bf16x16*>(Vp);
    acc = __builtin_amdgcn_wmma_f32_16x16x32_bf16(false, a, false, bfr,
                                                  (short)0, acc, false, false);
  }
#pragma unroll
  for (int r = 0; r < 8; ++r) {
    const int row = rbase + r;
    Y[((size_t)b * NT + q0 + row) * NC + h * ND + d0 + nn] = (__bf16)acc[r];
  }
}

// ---------------- output projection: [16384,768] @ Wt[768,768]^T + bias -> f32 ----------------
__global__ __launch_bounds__(32) void proj_gemm_kernel(const __bf16* __restrict__ Y,
                                                       const __bf16* __restrict__ WtP,
                                                       const float* __restrict__ b_proj,
                                                       float* __restrict__ out) {
  const int lane = threadIdx.x & 31;
  const int m0 = blockIdx.x * 16;
  const int n0 = blockIdx.y * 64;
  const int m    = lane & 15;
  const int nn   = lane & 15;
  const int kb_b = (lane >> 4) * 16;
  f32x8 acc[4] = {};
  const __bf16* Arow = Y + (size_t)(m0 + m) * NC;
  for (int k0 = 0; k0 < NC; k0 += 32) {
    bf16x16 a = load_a_frag(Arow + k0, lane);
#pragma unroll
    for (int t = 0; t < 4; ++t) {
      const __bf16* Bp = WtP + (size_t)(n0 + t * 16 + nn) * NC + k0 + kb_b;
      bf16x16 bfr = *reinterpret_cast<const bf16x16*>(Bp);
      acc[t] = __builtin_amdgcn_wmma_f32_16x16x32_bf16(false, a, false, bfr,
                                                       (short)0, acc[t], false, false);
    }
  }
  const int rbase = (lane >> 4) * 8;
#pragma unroll
  for (int t = 0; t < 4; ++t) {
    const int c = n0 + t * 16 + nn;
    const float bias = b_proj[c];
#pragma unroll
    for (int r = 0; r < 8; ++r) {
      const int row = m0 + rbase + r;
      out[(size_t)row * NC + c] = acc[t][r] + bias;
    }
  }
}

extern "C" void kernel_launch(void* const* d_in, const int* in_sizes, int n_in,
                              void* d_out, int out_size, void* d_ws, size_t ws_size,
                              hipStream_t stream) {
  (void)in_sizes; (void)n_in; (void)out_size; (void)ws_size;
  const float* x      = (const float*)d_in[0];
  const int*   amask  = (const int*)d_in[1];
  const float* W_attn = (const float*)d_in[2];
  const float* b_attn = (const float*)d_in[3];
  const float* W_proj = (const float*)d_in[4];
  const float* b_proj = (const float*)d_in[5];
  float* out = (float*)d_out;

  // workspace carve-up (all sizes are multiples of 256 bytes); total ~130.5 MB
  char* ws = (char*)d_ws;
  size_t off = 0;
  auto carve = [&](size_t bytes) {
    void* p = ws + off;
    off += (bytes + 255) & ~(size_t)255;
    return p;
  };
  const size_t n_x   = (size_t)NB * NT * NC;              // 12,582,912
  const size_t n_qkv = (size_t)NB * NH * NT * ND;         // 12,582,912 per tensor
  __bf16* xbf = (__bf16*)carve(n_x * 2);
  __bf16* WtA = (__bf16*)carve((size_t)3 * NC * NC * 2);  // [2304][768]
  __bf16* WtP = (__bf16*)carve((size_t)NC * NC * 2);      // [768][768]
  __bf16* Q   = (__bf16*)carve(n_qkv * 2);                // [B,H,T,D]
  __bf16* Kc  = (__bf16*)carve(n_qkv * 2);                // [B,H,T,D]
  __bf16* Vt  = (__bf16*)carve(n_qkv * 2);                // [B,H,D,T]
  __bf16* Ybf = (__bf16*)carve(n_x * 2);                  // [B*T, C]

  // 1) precision/layout prep
  cvt_bf16_kernel<<<(unsigned)((n_x + 255) / 256), 256, 0, stream>>>(x, xbf, (int)n_x);
  transpose_cvt_kernel<<<(unsigned)(((size_t)NC * 3 * NC + 255) / 256), 256, 0, stream>>>(
      W_attn, WtA, NC, 3 * NC);
  transpose_cvt_kernel<<<(unsigned)(((size_t)NC * NC + 255) / 256), 256, 0, stream>>>(
      W_proj, WtP, NC, NC);

  // 2) QKV GEMM: M=16384 (1024 tiles), N=2304 (36 tiles)
  qkv_gemm_kernel<<<dim3(NB * NT / 16, 3 * NC / 64), 32, 0, stream>>>(
      xbf, WtA, b_attn, Q, Kc, Vt);

  // 3) attention: (64 q-tiles, 12 heads, 16 batches), 4 waves per block
  attn_kernel<<<dim3(NT / 16, NH, NB), 128, 0, stream>>>(Q, Kc, Vt, amask, Ybf);

  // 4) output projection: M=16384, N=768 (12 tiles)
  proj_gemm_kernel<<<dim3(NB * NT / 16, NC / 64), 32, 0, stream>>>(
      Ybf, WtP, b_proj, out);
}
